// SNSelfAttention_54374285967636
// MI455X (gfx1250) — matrix operand order repaired and checked by reading, not compiled, not run on hardware
//
#include <hip/hip_runtime.h>
#include <math.h>

typedef __attribute__((ext_vector_type(16))) __bf16 v16bf;
typedef __attribute__((ext_vector_type(8)))  __bf16 v8bf;
typedef __attribute__((ext_vector_type(4)))  __bf16 v4bf;
typedef __attribute__((ext_vector_type(8)))  float  v8f;
typedef __attribute__((ext_vector_type(4)))  float  v4f;
typedef __attribute__((ext_vector_type(4)))  int    v4i;

// ---------- CDNA5 feature probes (device pass only) ----------
#if defined(__HIP_DEVICE_COMPILE__) && __has_builtin(__builtin_amdgcn_global_load_async_to_lds_b128)
  #define HAVE_ASYNC 1
#else
  #define HAVE_ASYNC 0
#endif
#if defined(__HIP_DEVICE_COMPILE__) && __has_builtin(__builtin_amdgcn_s_wait_asynccnt)
  #define WAIT_ASYNC() __builtin_amdgcn_s_wait_asynccnt(0)
#else
  #define WAIT_ASYNC() asm volatile("s_wait_asynccnt 0x0" ::: "memory")
#endif

#if HAVE_ASYNC
// async 16-byte global -> LDS copy (ASYNCcnt-tracked, no VGPR data)
__device__ __forceinline__ void async_cp16(const void* g, void* l) {
    auto gp = (__attribute__((address_space(1))) v4i*)(uintptr_t)g;
    auto lp = (__attribute__((address_space(3))) v4i*)(uintptr_t)l;
    __builtin_amdgcn_global_load_async_to_lds_b128(gp, lp, 0, 0);
}
#endif

// ---------- conversions: use the compiler's native bf16 lowering ----------
__device__ __forceinline__ __bf16 f2bf(float f) { return (__bf16)f; }
__device__ __forceinline__ float bf2f(__bf16 b) { return (float)b; }

// A-fragment: 16x32 bf16, lane M = lane&15, K = {8h..8h+7} U {16+8h..16+8h+7}
__device__ __forceinline__ v16bf load_frag_a(const __bf16* row, int half) {
    v8bf lo = *(const v8bf*)(row + half * 8);
    v8bf hi = *(const v8bf*)(row + 16 + half * 8);
    return __builtin_shufflevector(lo, hi, 0,1,2,3,4,5,6,7,8,9,10,11,12,13,14,15);
}
// B-fragment: 32x16 bf16, lane N = lane&15, K = {8h..8h+7} U {16+8h..16+8h+7}
__device__ __forceinline__ v16bf load_frag_b(const __bf16* row, int half) {
    v8bf lo = *(const v8bf*)(row + half * 8);
    v8bf hi = *(const v8bf*)(row + 16 + half * 8);
    return __builtin_shufflevector(lo, hi, 0,1,2,3,4,5,6,7,8,9,10,11,12,13,14,15);
}
#define WMMA_BF16(A, B, C) \
    __builtin_amdgcn_wmma_f32_16x16x32_bf16(false, (A), false, (B), (short)0, (C), false, false)

__device__ __forceinline__ float block_reduce_sum(float v, float* red) {
    #pragma unroll
    for (int off = 16; off; off >>= 1) v += __shfl_xor(v, off, 32);
    __syncthreads();
    if ((threadIdx.x & 31) == 0) red[threadIdx.x >> 5] = v;
    __syncthreads();
    float s = 0.f;
    #pragma unroll
    for (int i = 0; i < 8; ++i) s += red[i];
    return s;
}

// ---------- Kernel 1: spectral normalization (4 blocks, one per weight) ----------
__global__ __launch_bounds__(256) void sn_kernel(
    const float* __restrict__ w0, const float* __restrict__ u0,
    const float* __restrict__ w1, const float* __restrict__ u1,
    const float* __restrict__ w2, const float* __restrict__ u2,
    const float* __restrict__ w3, const float* __restrict__ u3,
    __bf16* __restrict__ wcat, __bf16* __restrict__ wa)
{
    __shared__ float uS[512];
    __shared__ float vS[512];
    __shared__ float red[8];
    const int which = blockIdx.x, tid = threadIdx.x;
    const float* Wp = (which == 0) ? w0 : (which == 1) ? w1 : (which == 2) ? w2 : w3;
    const float* Up = (which == 0) ? u0 : (which == 1) ? u1 : (which == 2) ? u2 : u3;
    const int R = (which == 3) ? 256 : 512;
    const int S = (which <= 1) ? 64 : (which == 2) ? 256 : 512;
    __bf16* dst = (which == 3) ? wa : (wcat + ((which == 0) ? 0 : (which == 1) ? 64 : 128));
    const int ldd = (which == 3) ? 512 : 384;

    for (int s = tid; s < S; s += 256) uS[s] = Up[s];
    __syncthreads();
    // v_un = W @ u
    for (int r = tid; r < R; r += 256) {
        float a = 0.f;
        for (int s = 0; s < S; ++s) a += Wp[(size_t)r * S + s] * uS[s];
        vS[r] = a;
    }
    __syncthreads();
    float part = 0.f;
    for (int r = tid; r < R; r += 256) part += vS[r] * vS[r];
    float n2 = block_reduce_sum(part, red);
    float inv = rsqrtf(fmaxf(n2, 1e-12f));
    for (int r = tid; r < R; r += 256) vS[r] *= inv;
    __syncthreads();
    // u2_un = W^T @ v
    for (int s = tid; s < S; s += 256) {
        float a = 0.f;
        for (int r = 0; r < R; ++r) a += Wp[(size_t)r * S + s] * vS[r];
        uS[s] = a;
    }
    __syncthreads();
    part = 0.f;
    for (int s = tid; s < S; s += 256) part += uS[s] * uS[s];
    float n2b = block_reduce_sum(part, red);
    // sigma = ||u2_un||^2 / sqrt(max(||u2_un||^2, eps))
    float sigma = n2b * rsqrtf(fmaxf(n2b, 1e-12f));
    float scale = 1.0f / sigma;
    for (int e = tid; e < R * S; e += 256) {
        int r = e / S, s = e - r * S;
        dst[(size_t)r * ldd + s] = f2bf(Wp[e] * scale);
    }
}

// ---------- Kernel 2/5: tiled GEMM, C[M,N] = A[M,K] * B[K,N], bf16 WMMA ----------
__global__ __launch_bounds__(256) void gemm_kernel(
    const void* __restrict__ A, int aIsF32, int lda,
    const __bf16* __restrict__ B, int ldb, int K,
    void* __restrict__ Cout, int outIsBF16, int ldc,
    const float* __restrict__ residual, const float* __restrict__ sigmaPtr)
{
    __shared__ __align__(16) __bf16 aS[128 * 32];
    __shared__ __align__(16) __bf16 bS[128 * 32];
    const int tid = threadIdx.x;
    const int wave = tid >> 5, lane = tid & 31, half = lane >> 4, l16 = lane & 15;
    const int row0 = blockIdx.y * 128, n0 = blockIdx.x * 128;

    v8f acc[8];
    #pragma unroll
    for (int nt = 0; nt < 8; ++nt) acc[nt] = (v8f)(0.0f);

    for (int k0 = 0; k0 < K; k0 += 32) {
        // stage A tile (128 x 32) as bf16
        {
            int r = tid >> 1;
            int kk = (tid & 1) * 16;
            if (aIsF32) {
                const float* src = (const float*)A + (size_t)(row0 + r) * lda + k0 + kk;
                #pragma unroll
                for (int i = 0; i < 4; ++i) {
                    v4f x = *(const v4f*)(src + 4 * i);
                    *(v4bf*)&aS[r * 32 + kk + 4 * i] = __builtin_convertvector(x, v4bf);
                }
            } else {
                const __bf16* src = (const __bf16*)A + (size_t)(row0 + r) * lda + k0 + kk;
#if HAVE_ASYNC
                async_cp16(src,     &aS[r * 32 + kk]);
                async_cp16(src + 8, &aS[r * 32 + kk + 8]);
#else
                #pragma unroll
                for (int i = 0; i < 16; ++i) aS[r * 32 + kk + i] = src[i];
#endif
            }
        }
        // stage B tile transposed: bS[n][k]
        #pragma unroll
        for (int i = 0; i < 16; ++i) {
            int e = tid * 16 + i;
            int n = e & 127, k = e >> 7;
            bS[n * 32 + k] = B[(size_t)(k0 + k) * ldb + n0 + n];
        }
#if HAVE_ASYNC
        WAIT_ASYNC();
#endif
        __syncthreads();

        v16bf af = load_frag_a(&aS[(wave * 16 + l16) * 32], half);
        #pragma unroll
        for (int nt = 0; nt < 8; ++nt) {
            v16bf bfr = load_frag_b(&bS[(nt * 16 + l16) * 32], half);
            acc[nt] = WMMA_BF16(af, bfr, acc[nt]);
        }
        __syncthreads();
    }

    float sig = sigmaPtr ? *sigmaPtr : 0.0f;
    #pragma unroll
    for (int nt = 0; nt < 8; ++nt) {
        #pragma unroll
        for (int r = 0; r < 8; ++r) {
            int grow = row0 + wave * 16 + r + 8 * half;
            int gcol = n0 + nt * 16 + l16;
            float v = acc[nt][r];
            if (residual) v = residual[(size_t)grow * ldc + gcol] + sig * v;
            if (outIsBF16) ((__bf16*)Cout)[(size_t)grow * ldc + gcol] = f2bf(v);
            else           ((float*)Cout)[(size_t)grow * ldc + gcol] = v;
        }
    }
}

// ---------- Kernel 3: 2x2 maxpool of P cols 64..383 -> phi, gT ----------
__global__ __launch_bounds__(256) void pool_kernel(
    const __bf16* __restrict__ Pbf, __bf16* __restrict__ phi, __bf16* __restrict__ gT)
{
    const int PHI_TOT = 8 * 1024 * 64;      // 524288
    const int G_TOT   = 8 * 256 * 1024;     // 2097152
    int idx = blockIdx.x * 256 + threadIdx.x;
    if (idx < PHI_TOT) {
        int b = idx >> 16;
        int rem = idx & 65535;
        int mo = rem >> 6, d = rem & 63;
        int h2 = mo >> 5, w2 = mo & 31;
        size_t base = ((size_t)b * 4096 + (size_t)(h2 * 2) * 64 + w2 * 2) * 384 + 64 + d;
        float v0 = bf2f(Pbf[base]),            v1 = bf2f(Pbf[base + 384]);
        float v2 = bf2f(Pbf[base + 64 * 384]), v3 = bf2f(Pbf[base + 65 * 384]);
        phi[idx] = f2bf(fmaxf(fmaxf(v0, v1), fmaxf(v2, v3)));
    } else if (idx < PHI_TOT + G_TOT) {
        int j = idx - PHI_TOT;
        int b = j >> 18;
        int rem = j & 262143;
        int d = rem >> 10, mo = rem & 1023;
        int h2 = mo >> 5, w2 = mo & 31;
        size_t base = ((size_t)b * 4096 + (size_t)(h2 * 2) * 64 + w2 * 2) * 384 + 128 + d;
        float v0 = bf2f(Pbf[base]),            v1 = bf2f(Pbf[base + 384]);
        float v2 = bf2f(Pbf[base + 64 * 384]), v3 = bf2f(Pbf[base + 65 * 384]);
        gT[j] = f2bf(fmaxf(fmaxf(v0, v1), fmaxf(v2, v3)));   // gT[b][d][mo]
    }
}

// ---------- Kernel 4: flash attention: AG = softmax(theta phi^T) g ----------
// 128 threads = 4 waves; wave w owns 16 query rows. Key blocks of 32, m=1024.
__global__ __launch_bounds__(128) void attn_kernel(
    const __bf16* __restrict__ Pbf,   // theta at cols 0..63 of [32768][384]
    const __bf16* __restrict__ phi,   // [8][1024][64]
    const __bf16* __restrict__ gT,    // [8][256][1024]
    __bf16* __restrict__ AG)          // [8][4096][256]
{
    __shared__ __align__(16) __bf16 qS[64 * 64];
    __shared__ __align__(16) __bf16 kS[32 * 64];
    __shared__ __align__(16) __bf16 gS[256 * 32];
    __shared__ __align__(16) __bf16 pS[4 * 16 * 32];
    const int b = blockIdx.y, qb = blockIdx.x;
    const int tid = threadIdx.x;
    const int wave = tid >> 5, lane = tid & 31, half = lane >> 4, l16 = lane & 15;

    // load 64x64 query tile (theta); per-thread 4 aligned 16B chunks
#if HAVE_ASYNC
    #pragma unroll
    for (int c = 0; c < 4; ++c) {
        int e0 = tid * 32 + c * 8;
        async_cp16(&Pbf[(size_t)(b * 4096 + qb * 64 + (e0 >> 6)) * 384 + (e0 & 63)], &qS[e0]);
    }
    WAIT_ASYNC();
#else
    #pragma unroll
    for (int i = 0; i < 32; ++i) {
        int e = tid * 32 + i;
        qS[e] = Pbf[(size_t)(b * 4096 + qb * 64 + (e >> 6)) * 384 + (e & 63)];
    }
#endif
    __syncthreads();

    float m_i[8], l_i[8];
    v8f acc[16];
    #pragma unroll
    for (int r = 0; r < 8; ++r) { m_i[r] = -INFINITY; l_i[r] = 0.f; }
    #pragma unroll
    for (int t = 0; t < 16; ++t) acc[t] = (v8f)(0.0f);

    for (int kb = 0; kb < 32; ++kb) {
        __syncthreads();
        // stage 32 keys x 64 d  and  gT tile 256 d x 32 keys
#if HAVE_ASYNC
        #pragma unroll
        for (int c = 0; c < 2; ++c) {
            int e0 = tid * 16 + c * 8;
            async_cp16(&phi[((size_t)b * 1024 + kb * 32 + (e0 >> 6)) * 64 + (e0 & 63)], &kS[e0]);
        }
        #pragma unroll
        for (int c = 0; c < 8; ++c) {
            int e0 = tid * 64 + c * 8;
            async_cp16(&gT[((size_t)b * 256 + (e0 >> 5)) * 1024 + kb * 32 + (e0 & 31)], &gS[e0]);
        }
        WAIT_ASYNC();
#else
        #pragma unroll
        for (int i = 0; i < 16; ++i) {
            int e = tid * 16 + i;
            kS[e] = phi[((size_t)b * 1024 + kb * 32 + (e >> 6)) * 64 + (e & 63)];
        }
        #pragma unroll
        for (int i = 0; i < 64; ++i) {
            int e = tid * 64 + i;
            gS[e] = gT[((size_t)b * 256 + (e >> 5)) * 1024 + kb * 32 + (e & 31)];
        }
#endif
        __syncthreads();

        // scores: 16 rows x 32 keys (two 16x16 tiles), K = d = 64
        v8f s0 = (v8f)(0.0f), s1 = (v8f)(0.0f);
        const __bf16* aRow  = &qS[(wave * 16 + l16) * 64];
        const __bf16* b0Row = &kS[l16 * 64];
        const __bf16* b1Row = &kS[(16 + l16) * 64];
        #pragma unroll
        for (int kk = 0; kk < 64; kk += 32) {
            v16bf af  = load_frag_a(aRow + kk, half);
            v16bf bf0 = load_frag_b(b0Row + kk, half);
            v16bf bf1 = load_frag_b(b1Row + kk, half);
            s0 = WMMA_BF16(af, bf0, s0);
            s1 = WMMA_BF16(af, bf1, s1);
        }

        // online softmax per row (row = r + 8*half, stats uniform over 16-lane group)
        float p0[8], p1[8];
        #pragma unroll
        for (int r = 0; r < 8; ++r) {
            float v = fmaxf(s0[r], s1[r]);
            #pragma unroll
            for (int off = 8; off; off >>= 1) v = fmaxf(v, __shfl_xor(v, off, 32));
            float mnew  = fmaxf(m_i[r], v);
            float scale = __expf(m_i[r] - mnew);
            p0[r] = __expf(s0[r] - mnew);
            p1[r] = __expf(s1[r] - mnew);
            float ps = p0[r] + p1[r];
            #pragma unroll
            for (int off = 8; off; off >>= 1) ps += __shfl_xor(ps, off, 32);
            l_i[r] = l_i[r] * scale + ps;
            m_i[r] = mnew;
            #pragma unroll
            for (int t = 0; t < 16; ++t) acc[t][r] *= scale;
        }

        // stage P (bf16) in A-fragment source layout [row][key]
        __bf16* pw = &pS[wave * 16 * 32];
        #pragma unroll
        for (int r = 0; r < 8; ++r) {
            pw[(r + 8 * half) * 32 + l16]      = f2bf(p0[r]);
            pw[(r + 8 * half) * 32 + 16 + l16] = f2bf(p1[r]);
        }
        __syncthreads();

        // O += P @ g : 16 output d-tiles, K = 32 keys
        v16bf pa = load_frag_a(&pS[(wave * 16 + l16) * 32], half);
        #pragma unroll
        for (int dt = 0; dt < 16; ++dt) {
            v16bf bg = load_frag_b(&gS[(dt * 16 + l16) * 32], half);
            acc[dt] = WMMA_BF16(pa, bg, acc[dt]);
        }
    }

    float invr[8];
    #pragma unroll
    for (int r = 0; r < 8; ++r) invr[r] = 1.0f / l_i[r];
    #pragma unroll
    for (int dt = 0; dt < 16; ++dt) {
        #pragma unroll
        for (int r = 0; r < 8; ++r) {
            size_t row = (size_t)b * 4096 + qb * 64 + wave * 16 + r + 8 * half;
            AG[row * 256 + dt * 16 + l16] = f2bf(acc[dt][r] * invr[r]);
        }
    }
}

// ---------- host launcher ----------
extern "C" void kernel_launch(void* const* d_in, const int* in_sizes, int n_in,
                              void* d_out, int out_size, void* d_ws, size_t ws_size,
                              hipStream_t stream) {
    (void)in_sizes; (void)n_in; (void)out_size; (void)ws_size;
    const float* X       = (const float*)d_in[0];
    const float* w_theta = (const float*)d_in[1];
    const float* u_theta = (const float*)d_in[2];
    const float* w_phi   = (const float*)d_in[3];
    const float* u_phi   = (const float*)d_in[4];
    const float* w_g     = (const float*)d_in[5];
    const float* u_g     = (const float*)d_in[6];
    const float* w_attn  = (const float*)d_in[7];
    const float* u_attn  = (const float*)d_in[8];
    const float* sigma   = (const float*)d_in[9];

    char* ws = (char*)d_ws;
    __bf16* wcat = (__bf16*)(ws);                         // 512*384*2   = 393216 B
    __bf16* wa   = (__bf16*)(ws + 393216);                // 256*512*2   = 262144 B
    __bf16* Pbf  = (__bf16*)(ws + 655360);                // 32768*384*2 = 25165824 B
    __bf16* phi  = (__bf16*)(ws + 25821184);              // 8*1024*64*2 = 1048576 B
    __bf16* gT   = (__bf16*)(ws + 26869760);              // 8*256*1024*2= 4194304 B
    __bf16* AG   = (__bf16*)(ws + 31064064);              // 32768*256*2 = 16777216 B
    // total workspace: 47841280 B (~46 MB)

    // 1) spectral norm -> normalized bf16 weights
    sn_kernel<<<4, 256, 0, stream>>>(w_theta, u_theta, w_phi, u_phi,
                                     w_g, u_g, w_attn, u_attn, wcat, wa);
    // 2) P = X @ [Wt|Wp|Wg]  (32768 x 384, bf16 out)
    gemm_kernel<<<dim3(3, 256), 256, 0, stream>>>(
        X, 1, 512, wcat, 384, 512, Pbf, 1, 384, nullptr, nullptr);
    // 3) 2x2 maxpool -> phi [8,1024,64], gT [8,256,1024]
    pool_kernel<<<(2621440 + 255) / 256, 256, 0, stream>>>(Pbf, phi, gT);
    // 4) flash attention -> AG [32768, 256] bf16
    attn_kernel<<<dim3(64, 8), 128, 0, stream>>>(Pbf, phi, gT, AG);
    // 5) out = X + sigma * (AG @ Wa)  (f32 out)
    gemm_kernel<<<dim3(4, 256), 256, 0, stream>>>(
        AG, 0, 256, wa, 512, 256, d_out, 0, 512, X, sigma);
}
